// GAT_ANE__72035191488868
// MI455X (gfx1250) — compile-verified, matrix operand order checked by reading, not run on hardware
//
#include <hip/hip_runtime.h>
#include <hip/hip_bf16.h>

typedef __attribute__((ext_vector_type(16))) _Float16 v16h;
typedef __attribute__((ext_vector_type(8)))  float    v8f;

#define NNODES 100000
#define NEDGES 1600000
#define NGRAPH 64
#define EPSF   1e-6f

// ---- ordered-uint encoding for float atomic max ----
__device__ __forceinline__ unsigned f2ord(float f) {
  unsigned u = __float_as_uint(f);
  return (u & 0x80000000u) ? ~u : (u | 0x80000000u);
}
__device__ __forceinline__ float ord2f(unsigned u) {
  unsigned b = (u & 0x80000000u) ? (u & 0x7fffffffu) : ~u;
  return __uint_as_float(b);
}

// ---- pack fw ([2*DI, DO] row-major) into WMMA B-fragment order as f16 ----
// Logical matrix W2[DI, 2*DO]: cols 0..DO-1 = fw[0:DI,:], cols DO..2DO-1 = fw[DI:2DI,:]
// Fragment layout (16x16x32 f16 B, 32xK16 tile): lane L holds col n = ct*16+(L&15),
// elems j<8 -> K = kt*32 + (L>>4)*8 + j ; elems j>=8 -> K = kt*32 + 16 + (L>>4)*8 + (j-8)
template<int DI, int DO>
__global__ void pack_w(const float* __restrict__ fw, _Float16* __restrict__ Bpk) {
  constexpr int KT = DI / 32;
  constexpr int CT = (2 * DO) / 16;
  int id = blockIdx.x * blockDim.x + threadIdx.x;
  if (id >= KT * CT * 512) return;
  int tile = id >> 9;
  int lane = (id >> 4) & 31;
  int j    = id & 15;
  int kt = tile / CT, ct = tile - kt * CT;
  int n = ct * 16 + (lane & 15);
  int half = lane >> 4;
  int k = kt * 32 + ((j < 8) ? (half * 8 + j) : (16 + half * 8 + (j - 8)));
  float v = (n < DO) ? fw[(size_t)k * DO + n] : fw[(size_t)(DI + k) * DO + (n - DO)];
  Bpk[id] = (_Float16)v;
}

// ---- attention GEMV: acat[2n]=x[n]@ww_top + wb, acat[2n+1]=x[n]@ww_bot ----
template<int DI>
__global__ void gemv_a(const float* __restrict__ X, const float* __restrict__ ww,
                       const float* __restrict__ wb, float* __restrict__ acat, int nrows) {
  int n = blockIdx.x * blockDim.x + threadIdx.x;
  if (n >= nrows) return;
  const float* xr = X + (size_t)n * DI;
  float s = 0.f, t = 0.f;
#pragma unroll
  for (int k = 0; k < DI; ++k) {
    float xv = xr[k];
    s = fmaf(xv, ww[k], s);
    t = fmaf(xv, ww[DI + k], t);
  }
  acat[2 * n]     = s + wb[0];
  acat[2 * n + 1] = t;
}

// ---- node projection via v_wmma_f32_16x16x32_f16: Pcat[N, 2*DO] = X[N,DI] @ W2 ----
template<int DI, int DO>
__global__ __launch_bounds__(256)
void proj_wmma(const float* __restrict__ X, const _Float16* __restrict__ Bpk,
               float* __restrict__ P, int nrows) {
  constexpr int KT = DI / 32;
  constexpr int CT = (2 * DO) / 16;
  const int lane = threadIdx.x & 31;
  const int wave = threadIdx.x >> 5;
  const int tile = blockIdx.x * (blockDim.x >> 5) + wave;
  if (tile * 16 >= nrows) return;            // wave-uniform exit, EXEC stays full
  const int m = lane & 15;
  const int half = lane >> 4;
  const int row = tile * 16 + m;
  const bool rv = row < nrows;

  const v8f zero = {0.f, 0.f, 0.f, 0.f, 0.f, 0.f, 0.f, 0.f};
  v8f acc[CT];
#pragma unroll
  for (int ct = 0; ct < CT; ++ct) acc[ct] = zero;

#pragma unroll
  for (int kt = 0; kt < KT; ++kt) {
    const float* xr = X + (size_t)row * DI + kt * 32 + half * 8;
    float4 f0 = make_float4(0.f, 0.f, 0.f, 0.f), f1 = f0, f2 = f0, f3 = f0;
    if (rv) {
      f0 = *(const float4*)(xr + 0);
      f1 = *(const float4*)(xr + 4);
      f2 = *(const float4*)(xr + 16);
      f3 = *(const float4*)(xr + 20);
    }
    v16h a;
    a[0] = (_Float16)f0.x;  a[1] = (_Float16)f0.y;  a[2] = (_Float16)f0.z;  a[3] = (_Float16)f0.w;
    a[4] = (_Float16)f1.x;  a[5] = (_Float16)f1.y;  a[6] = (_Float16)f1.z;  a[7] = (_Float16)f1.w;
    a[8] = (_Float16)f2.x;  a[9] = (_Float16)f2.y;  a[10] = (_Float16)f2.z; a[11] = (_Float16)f2.w;
    a[12] = (_Float16)f3.x; a[13] = (_Float16)f3.y; a[14] = (_Float16)f3.z; a[15] = (_Float16)f3.w;
#pragma unroll
    for (int ct = 0; ct < CT; ++ct) {
      const v16h b = *(const v16h*)(Bpk + ((size_t)(kt * CT + ct) * 32 + lane) * 16);
      acc[ct] = __builtin_amdgcn_wmma_f32_16x16x32_f16(
          /*neg_a=*/false, a, /*neg_b=*/false, b,
          /*c_mod=*/(short)0, acc[ct], /*reuse_a=*/false, /*reuse_b=*/false);
    }
  }

#pragma unroll
  for (int ct = 0; ct < CT; ++ct) {
#pragma unroll
    for (int r = 0; r < 8; ++r) {
      int ro = tile * 16 + r + 8 * half;
      if (ro < nrows) P[(size_t)ro * (2 * DO) + ct * 16 + m] = acc[ct][r];
    }
  }
}

// ---- global max over edges of asrc[src]+atgt[tgt] ----
__global__ __launch_bounds__(256)
void edge_max(const int* __restrict__ src, const int* __restrict__ tgt,
              const float* __restrict__ acat, int E, unsigned* __restrict__ amax_u) {
  int e = blockIdx.x * blockDim.x + threadIdx.x;
  float m = -3.0e38f;
  if (e < E) {
    int s = src[e], t = tgt[e];
    m = acat[2 * s] + acat[2 * t + 1];
  }
#pragma unroll
  for (int o = 16; o > 0; o >>= 1) m = fmaxf(m, __shfl_xor(m, o, 32));
  if ((threadIdx.x & 31) == 0) atomicMax(amax_u, f2ord(m));
}

// ---- per-edge: w = exp(a - amax); num[tgt] += relu(Psrc+Ptgt+fb)*w; den[tgt] += w ----
template<int DO>
__global__ __launch_bounds__(256)
void edge_scatter(const int* __restrict__ src, const int* __restrict__ tgt,
                  const float* __restrict__ acat, const unsigned* __restrict__ amax_u,
                  const float* __restrict__ P, const float* __restrict__ fb,
                  float* __restrict__ num, float* __restrict__ den, int E) {
  constexpr int C4 = DO / 4;
  int gid = blockIdx.x * blockDim.x + threadIdx.x;
  int e = gid / C4;                 // C4 is a power of two -> shift
  int c = gid - e * C4;
  if (e >= E) return;
  int s = src[e], t = tgt[e];
  float amax = ord2f(*amax_u);
  float w = __expf(acat[2 * s] + acat[2 * t + 1] - amax);
  float4 p  = *(const float4*)(P + (size_t)s * (2 * DO) + c * 4);
  float4 q  = *(const float4*)(P + (size_t)t * (2 * DO) + DO + c * 4);
  float4 bv = *(const float4*)(fb + c * 4);
  float* o = num + (size_t)t * DO + c * 4;
  atomicAdd(o + 0, fmaxf(p.x + q.x + bv.x, 0.f) * w);
  atomicAdd(o + 1, fmaxf(p.y + q.y + bv.y, 0.f) * w);
  atomicAdd(o + 2, fmaxf(p.z + q.z + bv.z, 0.f) * w);
  atomicAdd(o + 3, fmaxf(p.w + q.w + bv.w, 0.f) * w);
  if (c == 0) atomicAdd(den + t, w);
}

// ---- x_out = num / (den + eps), in place ----
template<int DO>
__global__ void normalize_k(float* __restrict__ num, const float* __restrict__ den, int total) {
  int i = blockIdx.x * blockDim.x + threadIdx.x;
  if (i >= total) return;
  int n = i / DO;
  num[i] = num[i] / (den[n] + EPSF);
}

// ---- graph readout: LDS-accumulated segment sum into gbuf[64,64] ----
__global__ __launch_bounds__(256)
void graph_accum(const float* __restrict__ X, const int* __restrict__ gids,
                 float* __restrict__ gbuf, int nrows) {
  __shared__ float acc[NGRAPH * 64];
  for (int i = threadIdx.x; i < NGRAPH * 64; i += blockDim.x) acc[i] = 0.f;
  __syncthreads();
  for (int n = blockIdx.x * blockDim.x + threadIdx.x; n < nrows; n += gridDim.x * blockDim.x) {
    int g = gids[n];
    const float* xr = X + (size_t)n * 64;
#pragma unroll
    for (int j = 0; j < 64; ++j) atomicAdd(&acc[g * 64 + j], xr[j]);
  }
  __syncthreads();
  for (int i = threadIdx.x; i < NGRAPH * 64; i += blockDim.x) atomicAdd(&gbuf[i], acc[i]);
}

// ---- tiny MLP head: out[64,10] = relu(g@mw0+mb0)@mw1+mb1 ----
__global__ __launch_bounds__(256)
void mlp_head(const float* __restrict__ g, const float* __restrict__ mw0,
              const float* __restrict__ mb0, const float* __restrict__ mw1,
              const float* __restrict__ mb1, float* __restrict__ out) {
  __shared__ float h[NGRAPH * 32];
  for (int idx = threadIdx.x; idx < NGRAPH * 32; idx += blockDim.x) {
    int r = idx >> 5, c = idx & 31;
    float s = mb0[c];
#pragma unroll
    for (int k = 0; k < 64; ++k) s = fmaf(g[r * 64 + k], mw0[k * 32 + c], s);
    h[idx] = fmaxf(s, 0.f);
  }
  __syncthreads();
  for (int idx = threadIdx.x; idx < NGRAPH * 10; idx += blockDim.x) {
    int r = idx / 10, c = idx - r * 10;
    float s = mb1[c];
#pragma unroll
    for (int k = 0; k < 32; ++k) s = fmaf(h[r * 32 + k], mw1[k * 10 + c], s);
    out[idx] = s;
  }
}

// ---- one GAT layer ----
template<int DI, int DO>
static void run_layer(const float* xin, float* xout, const float* fw, const float* fb,
                      const float* ww, const float* wb, const int* src, const int* tgt,
                      float* Pcat, float* acat, float* den, unsigned* amax, _Float16* Bpk,
                      hipStream_t stream) {
  constexpr int KT = DI / 32;
  constexpr int CT = (2 * DO) / 16;
  constexpr int C4 = DO / 4;
  hipMemsetAsync(amax, 0, 4, stream);
  hipMemsetAsync(xout, 0, (size_t)NNODES * DO * sizeof(float), stream);
  hipMemsetAsync(den, 0, (size_t)NNODES * sizeof(float), stream);
  pack_w<DI, DO><<<(KT * CT * 512 + 255) / 256, 256, 0, stream>>>(fw, Bpk);
  gemv_a<DI><<<(NNODES + 255) / 256, 256, 0, stream>>>(xin, ww, wb, acat, NNODES);
  unsigned ntiles = (NNODES + 15) / 16;
  proj_wmma<DI, DO><<<(ntiles + 7) / 8, 256, 0, stream>>>(xin, Bpk, Pcat, NNODES);
  edge_max<<<(NEDGES + 255) / 256, 256, 0, stream>>>(src, tgt, acat, NEDGES, amax);
  unsigned sg = (unsigned)(((size_t)NEDGES * C4 + 255) / 256);
  edge_scatter<DO><<<sg, 256, 0, stream>>>(src, tgt, acat, amax, Pcat, fb, xout, den, NEDGES);
  normalize_k<DO><<<(NNODES * DO + 255) / 256, 256, 0, stream>>>(xout, den, NNODES * DO);
}

extern "C" void kernel_launch(void* const* d_in, const int* in_sizes, int n_in,
                              void* d_out, int out_size, void* d_ws, size_t ws_size,
                              hipStream_t stream) {
  (void)in_sizes; (void)n_in; (void)out_size; (void)ws_size;
  const float* x    = (const float*)d_in[0];
  const int*   src  = (const int*)d_in[1];
  const int*   tgt  = (const int*)d_in[2];
  const int*   gids = (const int*)d_in[3];
  const float* fw0 = (const float*)d_in[4],  *fb0 = (const float*)d_in[5];
  const float* ww0 = (const float*)d_in[6],  *wb0 = (const float*)d_in[7];
  const float* fw1 = (const float*)d_in[8],  *fb1 = (const float*)d_in[9];
  const float* ww1 = (const float*)d_in[10], *wb1 = (const float*)d_in[11];
  const float* fw2 = (const float*)d_in[12], *fb2 = (const float*)d_in[13];
  const float* ww2 = (const float*)d_in[14], *wb2 = (const float*)d_in[15];
  const float* mw0 = (const float*)d_in[16], *mb0 = (const float*)d_in[17];
  const float* mw1 = (const float*)d_in[18], *mb1 = (const float*)d_in[19];
  float* out = (float*)d_out;

  char* ws = (char*)d_ws;
  size_t off = 0;
  auto carve = [&](size_t bytes) -> void* {
    void* p = ws + off;
    off = (off + bytes + 255) & ~(size_t)255;
    return p;
  };
  float*     Pcat = (float*)carve((size_t)NNODES * 128 * sizeof(float));   // 51.2 MB (L2-resident)
  float*     numA = (float*)carve((size_t)NNODES * 64 * sizeof(float));    // 25.6 MB
  float*     numB = (float*)carve((size_t)NNODES * 64 * sizeof(float));    // 25.6 MB
  float*     den  = (float*)carve((size_t)NNODES * sizeof(float));
  float*     acat = (float*)carve((size_t)NNODES * 2 * sizeof(float));
  unsigned*  amax = (unsigned*)carve(256);
  _Float16*  Bpk  = (_Float16*)carve(16 * 512 * sizeof(_Float16));
  float*     gbuf = (float*)carve(NGRAPH * 64 * sizeof(float));

  // Layer 0: di=64 -> do=32 ; Layer 1: di=32 -> do=64 ; Layer 2: di=64 -> do=64
  run_layer<64, 32>(x,    numA, fw0, fb0, ww0, wb0, src, tgt, Pcat, acat, den, amax, Bpk, stream);
  run_layer<32, 64>(numA, numB, fw1, fb1, ww1, wb1, src, tgt, Pcat, acat, den, amax, Bpk, stream);
  run_layer<64, 64>(numB, numA, fw2, fb2, ww2, wb2, src, tgt, Pcat, acat, den, amax, Bpk, stream);

  hipMemsetAsync(gbuf, 0, NGRAPH * 64 * sizeof(float), stream);
  graph_accum<<<128, 256, 0, stream>>>(numA, gids, gbuf, NNODES);
  mlp_head<<<1, 256, 0, stream>>>(gbuf, mw0, mb0, mw1, mb1, out);
}
